// PositionAwareAttention_31284541784221
// MI455X (gfx1250) — matrix operand order repaired and checked by reading, not compiled, not run on hardware
//
#include <hip/hip_runtime.h>
#include <math.h>

#define D_IN    512
#define D_MODEL 256
#define BATCH   2
#define SEQ     512

typedef __attribute__((ext_vector_type(16))) __bf16 v16bf;
typedef __attribute__((ext_vector_type(8)))  float  v8f;
typedef __attribute__((ext_vector_type(2)))  float  v2f;

#if defined(__has_builtin)
#  if __has_builtin(__builtin_amdgcn_wmma_f32_16x16x4_f32)
#    define HAVE_WMMA_F32K4 1
#  endif
#  if __has_builtin(__builtin_amdgcn_tanhf)
#    define FAST_TANH(x) __builtin_amdgcn_tanhf(x)
#  elif __has_builtin(__builtin_amdgcn_tanh_f32)
#    define FAST_TANH(x) __builtin_amdgcn_tanh_f32(x)
#  endif
#endif
#ifndef FAST_TANH
#  define FAST_TANH(x) tanhf(x)
#endif

// ---------------------------------------------------------------------------
// Kernel 1: h = tanh(x @ Wx + bx)   [B*N=1024, D_MODEL=256], K=D_IN=512
// One wave -> one 16x16 C tile, bf16 WMMA 16x16x32, bias folded into C init.
// ---------------------------------------------------------------------------
__global__ __launch_bounds__(256) void k_proj_tanh(const float* __restrict__ x,
                                                   const float* __restrict__ Wx,
                                                   const float* __restrict__ bx,
                                                   float* __restrict__ h) {
  const int lane16 = threadIdx.x & 15;
  const int half   = (threadIdx.x >> 4) & 1;
  const int wave   = threadIdx.x >> 5;
  const int tile   = blockIdx.x * 8 + wave;     // 1024 tiles total
  const int m0     = (tile >> 4) << 4;          // 64 tiles along M
  const int n0     = (tile & 15) << 4;          // 16 tiles along N
  const int col    = n0 + lane16;
  const int row    = m0 + lane16;

  v8f c;
  const float bias = bx[col];
#pragma unroll
  for (int v = 0; v < 8; ++v) c[v] = bias;

  for (int k0 = 0; k0 < D_IN; k0 += 32) {
    v16bf a, bm;
#pragma unroll
    for (int v = 0; v < 8; ++v) {
      const int ka = k0 + 2 * v + (v >= 4 ? 8 : 0) + 8 * half;  // A lane layout
      a[2 * v]     = (__bf16)x[row * D_IN + ka];
      a[2 * v + 1] = (__bf16)x[row * D_IN + ka + 1];
      const int kb = k0 + 2 * v + (v >= 4 ? 8 : 0) + 8 * half;  // B lane layout
      bm[2 * v]     = (__bf16)Wx[kb * D_MODEL + col];
      bm[2 * v + 1] = (__bf16)Wx[(kb + 1) * D_MODEL + col];
    }
    c = __builtin_amdgcn_wmma_f32_16x16x32_bf16(false, a, false, bm,
                                                (short)0, c, false, false);
  }
#pragma unroll
  for (int v = 0; v < 8; ++v)
    h[(m0 + v + 8 * half) * D_MODEL + col] = FAST_TANH(c[v]);
}

// ---------------------------------------------------------------------------
// Kernel 2: per (b,i) row of scores + masked softmax.
//   s[j] = sum_d w[d] * tanh( (pos[b,i]-pos[b,j]) . Wp[:,d] + bp[d] )
// K=4 contraction -> V_WMMA_F32_16X16X4_F32 (16 j-rows x 16 d-cols per tile),
// bias folded into C; tanh + w-weighting in VALU; lane-reduce over d.
// One block per (b,i): 8 waves x 4 j-tiles = 32 tiles = 512 j.
// ---------------------------------------------------------------------------
__global__ __launch_bounds__(256) void k_scores_softmax(const float* __restrict__ pos,
                                                        const int*   __restrict__ mask,
                                                        const float* __restrict__ Wp,
                                                        const float* __restrict__ bp,
                                                        const float* __restrict__ w,
                                                        float* __restrict__ attn) {
  __shared__ float pos_s[SEQ * 4];
  __shared__ float pi_s[4];
  __shared__ float sc[SEQ];
  __shared__ float red[256];

  const int b   = blockIdx.x / SEQ;
  const int i   = blockIdx.x % SEQ;
  const int tid = threadIdx.x;

  for (int t = tid; t < SEQ * 4; t += 256) pos_s[t] = pos[b * SEQ * 4 + t];
  if (tid < 4) pi_s[tid] = pos[(b * SEQ + i) * 4 + tid];
  __syncthreads();

#if HAVE_WMMA_F32K4
  {
    const int lane16 = tid & 15;
    const int half   = (tid >> 4) & 1;
    const int wave   = tid >> 5;
    for (int r = 0; r < 4; ++r) {
      const int jt = (wave * 4 + r) * 16;
      const int j  = jt + lane16;
      float rel[4];
#pragma unroll
      for (int k = 0; k < 4; ++k) rel[k] = pi_s[k] - pos_s[j * 4 + k];
      // A (16x4 f32): lane = M row (j), VGPR v -> K = v + 2*half
      v2f a;
      a.x = rel[2 * half];
      a.y = rel[2 * half + 1];

      v8f acc;
#pragma unroll
      for (int v = 0; v < 8; ++v) acc[v] = 0.0f;

      for (int dc = 0; dc < D_MODEL; dc += 16) {
        const int col = dc + lane16;
        // B (4x16 f32): VGPR v -> K = v + 2*half, lane = N col
        v2f bm;
        bm.x = Wp[(2 * half) * D_MODEL + col];
        bm.y = Wp[(2 * half + 1) * D_MODEL + col];
        v8f c;
        const float bias = bp[col];
#pragma unroll
        for (int v = 0; v < 8; ++v) c[v] = bias;
        c = __builtin_amdgcn_wmma_f32_16x16x4_f32(false, a, false, bm,
                                                  (short)0, c, false, false);
        const float wv = w[col];
#pragma unroll
        for (int v = 0; v < 8; ++v) acc[v] += FAST_TANH(c[v]) * wv;
      }
      // reduce over the 16 N-lanes within each half-wave
#pragma unroll
      for (int v = 0; v < 8; ++v) {
        float s = acc[v];
        for (int off = 8; off > 0; off >>= 1) s += __shfl_xor(s, off, 16);
        acc[v] = s;
      }
      if (lane16 == 0) {
#pragma unroll
        for (int v = 0; v < 8; ++v) sc[jt + v + 8 * half] = acc[v];
      }
    }
  }
#else
  // Scalar fallback (no f32 K=4 WMMA available)
  for (int j = tid; j < SEQ; j += 256) {
    const float r0 = pi_s[0] - pos_s[j * 4 + 0];
    const float r1 = pi_s[1] - pos_s[j * 4 + 1];
    const float r2 = pi_s[2] - pos_s[j * 4 + 2];
    const float r3 = pi_s[3] - pos_s[j * 4 + 3];
    float s = 0.0f;
    for (int d = 0; d < D_MODEL; ++d) {
      const float z = r0 * Wp[d] + r1 * Wp[D_MODEL + d] +
                      r2 * Wp[2 * D_MODEL + d] + r3 * Wp[3 * D_MODEL + d] + bp[d];
      s += FAST_TANH(z) * w[d];
    }
    sc[j] = s;
  }
#endif
  __syncthreads();

  // masked softmax over the 512-long row; each thread owns j = tid, tid+256
  const int mrow = (b * SEQ + i) * SEQ;
  const int m0 = mask[mrow + tid];
  const int m1 = mask[mrow + tid + 256];
  const float ninf = -__builtin_inff();
  const float s0 = m0 ? sc[tid] : ninf;
  const float s1 = m1 ? sc[tid + 256] : ninf;

  red[tid] = fmaxf(s0, s1);
  __syncthreads();
  for (int st = 128; st > 0; st >>= 1) {
    if (tid < st) red[tid] = fmaxf(red[tid], red[tid + st]);
    __syncthreads();
  }
  const float rmax = red[0];
  __syncthreads();

  const float e0 = m0 ? __expf(s0 - rmax) : 0.0f;
  const float e1 = m1 ? __expf(s1 - rmax) : 0.0f;
  red[tid] = e0 + e1;
  __syncthreads();
  for (int st = 128; st > 0; st >>= 1) {
    if (tid < st) red[tid] += red[tid + st];
    __syncthreads();
  }
  const float inv = 1.0f / red[0];

  attn[mrow + tid]       = m0 ? e0 * inv : 0.0f;
  attn[mrow + tid + 256] = m1 ? e1 * inv : 0.0f;
}

// ---------------------------------------------------------------------------
// Kernel 3: output[b] = attn[b] @ h[b]   (M=512, K=512, N=256 per batch)
// bf16 WMMA 16x16x32, on-the-fly f32->bf16 conversion.
// ---------------------------------------------------------------------------
__global__ __launch_bounds__(256) void k_out_gemm(const float* __restrict__ attn,
                                                  const float* __restrict__ h,
                                                  float* __restrict__ out) {
  const int lane16 = threadIdx.x & 15;
  const int half   = (threadIdx.x >> 4) & 1;
  const int wave   = threadIdx.x >> 5;
  const int tile   = blockIdx.x * 8 + wave;   // 1024 tiles (512 per batch)
  const int bidx   = tile >> 9;
  const int t      = tile & 511;
  const int m0     = (t >> 4) << 4;           // 32 i-tiles
  const int n0     = (t & 15) << 4;           // 16 d-tiles
  const int col    = n0 + lane16;
  const int row    = m0 + lane16;

  const float* Ab = attn + bidx * SEQ * SEQ;
  const float* Hb = h + bidx * SEQ * D_MODEL;

  v8f c;
#pragma unroll
  for (int v = 0; v < 8; ++v) c[v] = 0.0f;

  for (int k0 = 0; k0 < SEQ; k0 += 32) {
    v16bf a, bm;
#pragma unroll
    for (int v = 0; v < 8; ++v) {
      const int ka = k0 + 2 * v + (v >= 4 ? 8 : 0) + 8 * half;
      a[2 * v]     = (__bf16)Ab[row * SEQ + ka];
      a[2 * v + 1] = (__bf16)Ab[row * SEQ + ka + 1];
      const int kb = k0 + 2 * v + (v >= 4 ? 8 : 0) + 8 * half;
      bm[2 * v]     = (__bf16)Hb[kb * D_MODEL + col];
      bm[2 * v + 1] = (__bf16)Hb[(kb + 1) * D_MODEL + col];
    }
    c = __builtin_amdgcn_wmma_f32_16x16x32_bf16(false, a, false, bm,
                                                (short)0, c, false, false);
  }
#pragma unroll
  for (int v = 0; v < 8; ++v)
    out[(bidx * SEQ + m0 + v + 8 * half) * D_MODEL + col] = c[v];
}

// ---------------------------------------------------------------------------
extern "C" void kernel_launch(void* const* d_in, const int* in_sizes, int n_in,
                              void* d_out, int out_size, void* d_ws, size_t ws_size,
                              hipStream_t stream) {
  const float* x    = (const float*)d_in[0];
  const float* pos  = (const float*)d_in[1];
  const int*   amsk = (const int*)d_in[2];
  const float* Wx   = (const float*)d_in[3];
  const float* bx   = (const float*)d_in[4];
  const float* Wp   = (const float*)d_in[5];
  const float* bp   = (const float*)d_in[6];
  const float* w    = (const float*)d_in[7];

  float* out  = (float*)d_out;                       // [B, N, D_MODEL]
  float* attn = out + BATCH * SEQ * D_MODEL;         // [B, N, N]
  float* h    = (float*)d_ws;                        // [B*N, D_MODEL] f32

  k_proj_tanh<<<128, 256, 0, stream>>>(x, Wx, bx, h);
  k_scores_softmax<<<BATCH * SEQ, 256, 0, stream>>>(pos, amsk, Wp, bp, w, attn);
  k_out_gemm<<<128, 256, 0, stream>>>(attn, h, out);
}
